// WindowAttentionBlock_36782099923651
// MI455X (gfx1250) — compile-verified, matrix operand order checked by reading, not run on hardware
//
#include <hip/hip_runtime.h>

typedef __attribute__((ext_vector_type(16))) _Float16 v16h;
typedef __attribute__((ext_vector_type(8)))  _Float16 v8h;
typedef __attribute__((ext_vector_type(8)))  float    v8f;
typedef __attribute__((ext_vector_type(4)))  unsigned int v4u;
typedef __attribute__((ext_vector_type(8)))  int      v8i;
typedef __attribute__((ext_vector_type(4)))  int      v4i;

#define WMMA_F16(A, B, C) \
  __builtin_amdgcn_wmma_f32_16x16x32_f16(false, (A), false, (B), (short)0, (C), false, false)

// Load a 16-lane A/B fragment for v_wmma_f32_16x16x32_f16.
// Per ISA layout: lane L -> row (L&15); k-values = kbase + (L>>4)*8 + {0..7}
// for vgprs 0..3 and kbase + 16 + (L>>4)*8 + {0..7} for vgprs 4..7.
__device__ __forceinline__ v16h load_frag(const _Float16* __restrict__ base,
                                          int row, int stride, int kbase, int half) {
  const _Float16* p = base + row * stride + kbase + half * 8;
  union { v16h v; v8h h[2]; } u;
  u.h[0] = *(const v8h*)(p);
  u.h[1] = *(const v8h*)(p + 16);
  return u.v;
}

// LayerNorm over C=192 for 64 tokens; 4 threads per token, shuffle-reduce.
__device__ __forceinline__ void layernorm64(const float* __restrict__ X,
                                            _Float16* __restrict__ D,
                                            const float* __restrict__ g,
                                            const float* __restrict__ b, int tid) {
  int t = tid >> 2, p = tid & 3;
  const float* row = X + t * 192;
  int c0 = p * 48;
  float4 vx[12];
  float s = 0.f, sq = 0.f;
#pragma unroll
  for (int i = 0; i < 12; ++i) {
    float4 v = *(const float4*)(row + c0 + i * 4);
    vx[i] = v;
    s  += v.x + v.y + v.z + v.w;
    sq += v.x * v.x + v.y * v.y + v.z * v.z + v.w * v.w;
  }
  s  += __shfl_xor(s, 1, 4);  s  += __shfl_xor(s, 2, 4);
  sq += __shfl_xor(sq, 1, 4); sq += __shfl_xor(sq, 2, 4);
  float mean = s * (1.f / 192.f);
  float var  = sq * (1.f / 192.f) - mean * mean;
  float inv  = rsqrtf(var + 1e-5f);
  _Float16* drow = D + t * 200;
#pragma unroll
  for (int i = 0; i < 12; ++i) {
    float4 gv = *(const float4*)(g + c0 + i * 4);
    float4 bv = *(const float4*)(b + c0 + i * 4);
    union { _Float16 h[4]; double d; } o;
    o.h[0] = (_Float16)((vx[i].x - mean) * inv * gv.x + bv.x);
    o.h[1] = (_Float16)((vx[i].y - mean) * inv * gv.y + bv.y);
    o.h[2] = (_Float16)((vx[i].z - mean) * inv * gv.z + bv.z);
    o.h[3] = (_Float16)((vx[i].w - mean) * inv * gv.w + bv.w);
    *(double*)(drow + c0 + i * 4) = o.d;
  }
}

// Weight convert f32 -> f16 with transpose: dst (N x K) <- src (K x N)
__global__ void wconv_kernel(const float* __restrict__ src, _Float16* __restrict__ dst,
                             int K, int N) {
  int i = blockIdx.x * blockDim.x + threadIdx.x;
  if (i >= K * N) return;
  int n = i / K, k = i - n * K;
  dst[i] = (_Float16)src[(size_t)k * N + n];
}

// TDM descriptor for one 8x1536-f32 window tile (2D), groups 2/3 zero.
__device__ __forceinline__ void make_tdm_desc(const void* gptr, v4u& g0, v8i& g1) {
  unsigned long long ga = (unsigned long long)gptr;
  g0.x = 1u;                                   // count=1, user mode
  g0.y = 0u;                                   // lds_addr = 0 (XW at LDS base)
  g0.z = (unsigned int)(ga & 0xffffffffu);     // global_addr[31:0]
  g0.w = (unsigned int)((ga >> 32) & 0x1ffffffu) | 0x80000000u;  // addr[56:32] | type=2
  g1[0] = 0x20000;                 // workgroup_mask=0 | data_size=2 (4 bytes)
  g1[1] = (int)(1536u << 16);      // tensor_dim0 = 1536 (lo16)
  g1[2] = (int)(8u << 16);         // tensor_dim0 hi=0 | tensor_dim1 = 8 (lo16)
  g1[3] = (int)(1536u << 16);      // tensor_dim1 hi=0 | tile_dim0 = 1536
  g1[4] = 8;                       // tile_dim1 = 8 | tile_dim2 = 0
  g1[5] = 49152;                   // tensor_dim0_stride = 256*192 elements (lo32)
  g1[6] = 0;                       // stride0 hi | tensor_dim1_stride lo
  g1[7] = 0;
}

// LDS layout (dynamic shared, 172032 bytes total):
//  XW : f32 [64][192]            49152 B   residual stream (xw -> xo -> final out)
//  HA : f16 [64][200]            25600 B   A-operand buffer (ln1 -> attn out -> ln2)
//  QB : f16 [64][200]            25600 B   q  (reused as fc1-chunk output in MLP)
//  KB : f16 [64][200]            25600 B   k
//  VT : f16 [192][72]            27648 B   v transposed (dim-major)
//  PB : f16 [8][16][72]          18432 B   per-wave softmax(P) re-layout pad
#define SMEM_BYTES 172032

__global__ __launch_bounds__(256)
void swin_block_kernel(const float* __restrict__ x,
                       const float* __restrict__ g1, const float* __restrict__ b1,
                       const float* __restrict__ bqkv, const float* __restrict__ bproj,
                       const float* __restrict__ g2, const float* __restrict__ b2,
                       const float* __restrict__ bfc1, const float* __restrict__ bfc2,
                       const _Float16* __restrict__ wqkvT,
                       const _Float16* __restrict__ wprojT,
                       const _Float16* __restrict__ wfc1T,
                       const _Float16* __restrict__ wfc2T,
                       float* __restrict__ out) {
  extern __shared__ __align__(16) char smem[];
  float*    XW = (float*)smem;
  _Float16* HA = (_Float16*)(smem + 49152);
  _Float16* QB = (_Float16*)(smem + 49152 + 25600);
  _Float16* KB = (_Float16*)(smem + 49152 + 51200);
  _Float16* VT = (_Float16*)(smem + 49152 + 76800);
  _Float16* PB = (_Float16*)(smem + 49152 + 104448);

  const int widx = blockIdx.x;           // 4096 windows
  const int wb = widx >> 10;             // image
  const int wy = (widx >> 5) & 31;       // window row
  const int wx = widx & 31;              // window col
  const int tid  = threadIdx.x;
  const int lane = tid & 31;
  const int wv   = tid >> 5;             // wave 0..7
  const int half = lane >> 4;
  const int ln   = lane & 15;
  // fixed row-block per wave -> A fragments hoisted & reused across tiles
  const int mtw  = wv & 3;               // M tile 0..3 (rows mtw*16..+15)
  const int ntg  = wv >> 2;              // 0/1: which half of the N tiles

  const size_t win_base =
      (((size_t)(wb * 256 + wy * 8) * 256) + (size_t)(wx * 8)) * 192;

  // ---- Phase 0: TDM window gather (8 rows x 1536 f32) into XW ----
  if (__builtin_amdgcn_readfirstlane(tid) == 0) {   // scalar branch: TDM ignores EXEC
    v4u g0; v8i g1d; v4i z4 = {0, 0, 0, 0}; v8i z8 = {0, 0, 0, 0, 0, 0, 0, 0};
    make_tdm_desc(x + win_base, g0, g1d);
    __builtin_amdgcn_tensor_load_to_lds(g0, g1d, z4, z4, z8, 0);
    __builtin_amdgcn_s_wait_tensorcnt((short)0);
  }
  __syncthreads();

  // ---- Phase 1: LN1 -> HA (f16) ----
  layernorm64(XW, HA, g1, b1, tid);
  __syncthreads();

  // ---- Phase 2: QKV = HA @ Wqkv  (M=64, K=192, N=576) ----
  {
    v16h Af[6];
#pragma unroll
    for (int kk = 0; kk < 6; ++kk)
      Af[kk] = load_frag(HA, mtw * 16 + ln, 200, kk * 32, half);
    for (int it = 0; it < 18; ++it) {
      int nt = ntg * 18 + it;
      v8f acc = {};
#pragma unroll
      for (int kk = 0; kk < 6; ++kk) {
        v16h B = load_frag(wqkvT, nt * 16 + ln, 192, kk * 32, half);
        acc = WMMA_F16(Af[kk], B, acc);
      }
      int ncol = nt * 16 + ln;
      float bias = bqkv[ncol];
      if (ncol < 192) {
#pragma unroll
        for (int j = 0; j < 8; ++j)
          QB[(mtw * 16 + j + half * 8) * 200 + ncol] = (_Float16)(acc[j] + bias);
      } else if (ncol < 384) {
        int c = ncol - 192;
#pragma unroll
        for (int j = 0; j < 8; ++j)
          KB[(mtw * 16 + j + half * 8) * 200 + c] = (_Float16)(acc[j] + bias);
      } else {
        int d = ncol - 384;
#pragma unroll
        for (int j = 0; j < 8; ++j)
          VT[d * 72 + mtw * 16 + j + half * 8] = (_Float16)(acc[j] + bias);
      }
    }
  }
  __syncthreads();

  // ---- Phase 3: attention. 24 tasks = 6 heads x 4 row-chunks; 3 per wave ----
  const float sc = 0.176776695f;  // 1/sqrt(32)
  for (int rep = 0; rep < 3; ++rep) {
    int task = wv + rep * 8;
    int hh = task >> 2;      // head
    int rc = task & 3;       // 16-row chunk
    v8f s[4];
    v16h Aq = load_frag(QB, rc * 16 + ln, 200, hh * 32, half);
#pragma unroll
    for (int nt = 0; nt < 4; ++nt) {
      v8f z = {};
      v16h Bk = load_frag(KB, nt * 16 + ln, 200, hh * 32, half);
      s[nt] = WMMA_F16(Aq, Bk, z);
    }
    // in-register softmax: row = (fixed vgpr j, fixed half); 64 cols live in
    // 16 lanes of the half across 4 tiles.
#pragma unroll
    for (int j = 0; j < 8; ++j) {
      float a0 = s[0][j] * sc, a1 = s[1][j] * sc, a2 = s[2][j] * sc, a3 = s[3][j] * sc;
      float mx = fmaxf(fmaxf(a0, a1), fmaxf(a2, a3));
#pragma unroll
      for (int off = 1; off < 16; off <<= 1) mx = fmaxf(mx, __shfl_xor(mx, off, 16));
      a0 = __expf(a0 - mx); a1 = __expf(a1 - mx);
      a2 = __expf(a2 - mx); a3 = __expf(a3 - mx);
      float sum = a0 + a1 + a2 + a3;
#pragma unroll
      for (int off = 1; off < 16; off <<= 1) sum += __shfl_xor(sum, off, 16);
      float inv = 1.f / sum;
      s[0][j] = a0 * inv; s[1][j] = a1 * inv; s[2][j] = a2 * inv; s[3][j] = a3 * inv;
    }
    // D-layout -> A-layout via per-wave LDS pad
    _Float16* pw = PB + wv * (16 * 72);
#pragma unroll
    for (int t = 0; t < 4; ++t)
#pragma unroll
      for (int j = 0; j < 8; ++j)
        pw[(j + half * 8) * 72 + t * 16 + ln] = (_Float16)s[t][j];
    asm volatile("s_wait_dscnt 0x0" ::: "memory");   // same-wave LDS RAW
    // O = P @ V  (M=16, N=32, K=64)
#pragma unroll
    for (int dt = 0; dt < 2; ++dt) {
      v8f o = {};
#pragma unroll
      for (int kk = 0; kk < 2; ++kk) {
        v16h A = load_frag(pw, ln, 72, kk * 32, half);
        v16h B = load_frag(VT, hh * 32 + dt * 16 + ln, 72, kk * 32, half);
        o = WMMA_F16(A, B, o);
      }
#pragma unroll
      for (int j = 0; j < 8; ++j)
        HA[(rc * 16 + j + half * 8) * 200 + hh * 32 + dt * 16 + ln] = (_Float16)o[j];
    }
  }
  __syncthreads();

  // ---- Phase 4: proj (64x192x192) + residual into XW ----
  {
    v16h Af[6];
#pragma unroll
    for (int kk = 0; kk < 6; ++kk)
      Af[kk] = load_frag(HA, mtw * 16 + ln, 200, kk * 32, half);
#pragma unroll
    for (int it = 0; it < 6; ++it) {
      int nt = ntg * 6 + it;
      v8f acc = {};
#pragma unroll
      for (int kk = 0; kk < 6; ++kk) {
        v16h B = load_frag(wprojT, nt * 16 + ln, 192, kk * 32, half);
        acc = WMMA_F16(Af[kk], B, acc);
      }
      int ncol = nt * 16 + ln;
      float bias = bproj[ncol];
#pragma unroll
      for (int j = 0; j < 8; ++j) {
        int m = mtw * 16 + j + half * 8;
        XW[m * 192 + ncol] += acc[j] + bias;
      }
    }
  }
  __syncthreads();

  // ---- Phase 5: LN2 -> HA ----
  layernorm64(XW, HA, g2, b2, tid);
  __syncthreads();

  // ---- Phase 6: MLP, hidden in 4 chunks of 192; fc2 accumulators in regs ----
  _Float16* M1 = QB;  // reuse q buffer for fc1 chunk output
  v8f h2[6] = {};
  {
    v16h Af[6];
#pragma unroll
    for (int kk = 0; kk < 6; ++kk)
      Af[kk] = load_frag(HA, mtw * 16 + ln, 200, kk * 32, half);
    for (int chunk = 0; chunk < 4; ++chunk) {
      // fc1 chunk + GELU(exact)
#pragma unroll
      for (int it = 0; it < 6; ++it) {
        int nt = ntg * 6 + it;           // hidden tile within chunk, 0..11
        v8f acc = {};
#pragma unroll
        for (int kk = 0; kk < 6; ++kk) {
          v16h B = load_frag(wfc1T, chunk * 192 + nt * 16 + ln, 192, kk * 32, half);
          acc = WMMA_F16(Af[kk], B, acc);
        }
        int hid = chunk * 192 + nt * 16 + ln;
        float bias = bfc1[hid];
#pragma unroll
        for (int j = 0; j < 8; ++j) {
          float xv = acc[j] + bias;
          float gl = 0.5f * xv * (1.f + erff(xv * 0.70710678f));
          M1[(mtw * 16 + j + half * 8) * 200 + nt * 16 + ln] = (_Float16)gl;
        }
      }
      __syncthreads();
      // fc2 partial accumulate; A fragments from M1 hoisted per chunk
      v16h Am[6];
#pragma unroll
      for (int kk = 0; kk < 6; ++kk)
        Am[kk] = load_frag(M1, mtw * 16 + ln, 200, kk * 32, half);
#pragma unroll
      for (int it = 0; it < 6; ++it) {
        int nt = ntg * 6 + it;
#pragma unroll
        for (int kk = 0; kk < 6; ++kk) {
          v16h B = load_frag(wfc2T, nt * 16 + ln, 768, chunk * 192 + kk * 32, half);
          h2[it] = WMMA_F16(Am[kk], B, h2[it]);
        }
      }
      __syncthreads();
    }
  }

  // ---- Phase 7: final = xo + h2 + bfc2 written into XW, then TDM store ----
#pragma unroll
  for (int it = 0; it < 6; ++it) {
    int nt = ntg * 6 + it;
    int ncol = nt * 16 + ln;
    float bias = bfc2[ncol];
#pragma unroll
    for (int j = 0; j < 8; ++j) {
      int m = mtw * 16 + j + half * 8;
      XW[m * 192 + ncol] += h2[it][j] + bias;
    }
  }
  __syncthreads();
  if (__builtin_amdgcn_readfirstlane(tid) == 0) {   // scalar branch: TDM ignores EXEC
    v4u g0; v8i g1d; v4i z4 = {0, 0, 0, 0}; v8i z8 = {0, 0, 0, 0, 0, 0, 0, 0};
    make_tdm_desc(out + win_base, g0, g1d);
    __builtin_amdgcn_tensor_store_from_lds(g0, g1d, z4, z4, z8, 0);
    __builtin_amdgcn_s_wait_tensorcnt((short)0);
    // wave0's s_endpgm also implicitly waits; LDS persists until then
  }
}

extern "C" void kernel_launch(void* const* d_in, const int* in_sizes, int n_in,
                              void* d_out, int out_size, void* d_ws, size_t ws_size,
                              hipStream_t stream) {
  (void)in_sizes; (void)n_in; (void)out_size; (void)ws_size;
  const float* x     = (const float*)d_in[0];
  const float* g1    = (const float*)d_in[1];
  const float* b1    = (const float*)d_in[2];
  const float* wqkv  = (const float*)d_in[3];
  const float* bqkv  = (const float*)d_in[4];
  const float* wproj = (const float*)d_in[5];
  const float* bproj = (const float*)d_in[6];
  const float* g2    = (const float*)d_in[7];
  const float* b2    = (const float*)d_in[8];
  const float* wfc1  = (const float*)d_in[9];
  const float* bfc1  = (const float*)d_in[10];
  const float* wfc2  = (const float*)d_in[11];
  const float* bfc2  = (const float*)d_in[12];

  _Float16* ws = (_Float16*)d_ws;
  _Float16* wqkvT  = ws;               // 576 x 192
  _Float16* wprojT = ws + 110592;      // 192 x 192
  _Float16* wfc1T  = ws + 147456;      // 768 x 192
  _Float16* wfc2T  = ws + 294912;      // 192 x 768

  wconv_kernel<<<(110592 + 255) / 256, 256, 0, stream>>>(wqkv, wqkvT, 192, 576);
  wconv_kernel<<<(36864 + 255) / 256, 256, 0, stream>>>(wproj, wprojT, 192, 192);
  wconv_kernel<<<(147456 + 255) / 256, 256, 0, stream>>>(wfc1, wfc1T, 192, 768);
  wconv_kernel<<<(147456 + 255) / 256, 256, 0, stream>>>(wfc2, wfc2T, 768, 192);

  swin_block_kernel<<<4096, 256, SMEM_BYTES, stream>>>(
      x, g1, b1, bqkv, bproj, g2, b2, bfc1, bfc2,
      wqkvT, wprojT, wfc1T, wfc2T, (float*)d_out);
}
